// MoEExpertCollection_58317065945538
// MI455X (gfx1250) — matrix operand order repaired and checked by reading, not compiled, not run on hardware
//
#include <hip/hip_runtime.h>
#include <hip/hip_fp16.h>

typedef __attribute__((ext_vector_type(4)))  _Float16 v4h;
typedef __attribute__((ext_vector_type(8)))  _Float16 v8h;
typedef __attribute__((ext_vector_type(16))) _Float16 v16h;
typedef __attribute__((ext_vector_type(8)))  float    v8f;

#define D_IN      1024
#define D_OUT     2048
#define NEXP      16
#define NTOK      65536
#define TOK_PER_E (NTOK / NEXP)   /* 4096 */

#define BM 128
#define BN 128
#define BK 32
#define LDA 40   /* padded LDS row stride (halves): 80B rows -> conflict-free b128 */
#define LDB 40
#define KTILES (D_IN / BK)        /* 32 */

/* pre-pass tile */
#define PK 64    /* k rows per block   */
#define PN 32    /* n cols per block   */
#define TS_LD 72 /* padded k-stride in Ts (halves) */

// ---------------------------------------------------------------------------
// Pre-pass: routing_matrix [K][N][E] fp32  ->  Wh [E][N][K] f16 (k-innermost),
// transposing through LDS so both global reads and writes are coalesced.
// ---------------------------------------------------------------------------
__global__ __launch_bounds__(256) void moe_convert_w_kernel(
    const float* __restrict__ W, _Float16* __restrict__ Wh) {
  const int n0 = blockIdx.x * PN;
  const int k0 = blockIdx.y * PK;
  __shared__ __align__(16) _Float16 Ts[8][PN][TS_LD];   // 36,864 B

  const int t  = threadIdx.x;
  const int tn = t & 31;         // n within tile
  const int tk = (t >> 5) * 8;   // k base: 0,8,...,56

  for (int g = 0; g < 2; ++g) {  // expert groups {0..7}, {8..15}
    v8h ve[8];                   // per-expert run of 8 consecutive k
#pragma unroll
    for (int kk = 0; kk < 8; ++kk) {
      const float4* src = (const float4*)(
          W + ((size_t)(k0 + tk + kk) * D_OUT + (n0 + tn)) * NEXP + g * 8);
      float4 v0 = src[0];
      float4 v1 = src[1];
      ve[0][kk] = (_Float16)v0.x;  ve[1][kk] = (_Float16)v0.y;
      ve[2][kk] = (_Float16)v0.z;  ve[3][kk] = (_Float16)v0.w;
      ve[4][kk] = (_Float16)v1.x;  ve[5][kk] = (_Float16)v1.y;
      ve[6][kk] = (_Float16)v1.z;  ve[7][kk] = (_Float16)v1.w;
    }
#pragma unroll
    for (int e = 0; e < 8; ++e)
      *(v8h*)&Ts[e][tn][tk] = ve[e];
    __syncthreads();

#pragma unroll
    for (int p = 0; p < 8; ++p) {            // 256 rows of (e,n), 32 per pass
      int row = p * 32 + (t >> 3);           // 0..255
      int e   = row >> 5;                    // 0..7
      int n   = row & 31;
      int kl  = (t & 7) * 8;                 // 8 lanes cover one 64-half row
      v8h hv  = *(const v8h*)&Ts[e][n][kl];
      *(v8h*)(Wh + ((size_t)(g * 8 + e) * D_OUT + n0 + n) * D_IN + k0 + kl) = hv;
    }
    __syncthreads();                          // Ts reused by next group
  }
}

// ---------------------------------------------------------------------------
// Main grouped-GEMM kernel. grid = (N/BN, M_per_expert/BM, NEXP), block = 256.
// WS=true : B tiles DMA'd from Wh [e][n][k] f16 with GLOBAL_LOAD_ASYNC_TO_LDS
//           (tracked by ASYNCcnt; each wave waits on its own issues before the
//           publishing barrier). One barrier per K-iteration.
// WS=false: B gathered directly from routing_matrix [k][n][e] fp32 (fallback).
// ---------------------------------------------------------------------------
template <bool WS>
__global__ __launch_bounds__(256) void moe_gemm_silu_kernel(
    const float*    __restrict__ A,      // [NTOK][D_IN] fp32 (sorted by expert)
    const float*    __restrict__ W32,    // [D_IN][D_OUT][NEXP] fp32 (WS=false)
    const _Float16* __restrict__ Wh,     // [NEXP][D_OUT][D_IN] f16  (WS=true)
    const float*    __restrict__ bias,   // [D_OUT][NEXP] fp32
    float*          __restrict__ out) {  // [NTOK][D_OUT] fp32
  const int e  = blockIdx.z;
  const int n0 = blockIdx.x * BN;
  const int m0 = blockIdx.y * BM;
  const size_t tokBase = (size_t)e * TOK_PER_E + m0;
  const float* Ag = A + tokBase * D_IN;

  __shared__ __align__(16) _Float16 As[2][BM * LDA];  // 20 KB
  __shared__ __align__(16) _Float16 Bs[2][BN * LDB];  // 20 KB

  const int t     = threadIdx.x;
  const int wave  = t >> 5;
  const int lane  = t & 31;
  const int wm    = wave & 1;    // 2 waves along M (64 rows each)
  const int wn    = wave >> 1;   // 4 waves along N (32 cols each)
  const int lmod  = lane & 15;
  const int lhalf = lane >> 4;

  // ---- staging ------------------------------------------------------------
  float4 aReg[4];                 // A: rows p*32 + t/8, cols (t&7)*4
  float  bRegF[16];               // B (gather fallback)
  const int arow = t >> 3;        // 0..31
  const int acol = (t & 7) * 4;   // 0..28
  const int brow = t >> 1;        // 0..127
  const int bcol = (t & 1) * 16;  // 0 or 16 (halves)

  auto fetchA = [&](int kt) {
    const float* src = Ag + kt * BK;
#pragma unroll
    for (int p = 0; p < 4; ++p)
      aReg[p] = *(const float4*)(src + (size_t)(p * 32 + arow) * D_IN + acol);
  };
  auto storeA = [&](int buf) {
#pragma unroll
    for (int p = 0; p < 4; ++p) {
      v4h h = { (_Float16)aReg[p].x, (_Float16)aReg[p].y,
                (_Float16)aReg[p].z, (_Float16)aReg[p].w };
      *(v4h*)&As[buf][(p * 32 + arow) * LDA + acol] = h;
    }
  };
  // WS path: async DMA 32B (2 x b128) straight into Bs[buf]; no VGPR staging.
  // INST_OFFSET is added to BOTH the memory and the LDS address (ISA 10.x),
  // so one base pair + offset:16 covers the second b128.
  auto issueB = [&](int kt, int buf) {
    const _Float16* src = Wh + ((size_t)e * D_OUT + n0 + brow) * D_IN
                             + kt * BK + bcol;
    unsigned ldsOff = (unsigned)(size_t)&Bs[buf][brow * LDB + bcol];  // low 32b = LDS offset
    asm volatile("global_load_async_to_lds_b128 %0, %1, off"
                 :: "v"(ldsOff), "v"(src) : "memory");
    asm volatile("global_load_async_to_lds_b128 %0, %1, off offset:16"
                 :: "v"(ldsOff), "v"(src) : "memory");
  };
  auto fetchB = [&](int kt) {   // fallback gather
    const int k  = t & 31;
    const int nb = t >> 5;
#pragma unroll
    for (int j = 0; j < 16; ++j)
      bRegF[j] = W32[(((size_t)(kt * BK + k)) * D_OUT + n0 + nb + j * 8) * NEXP + e];
  };
  auto storeB = [&](int buf) {  // fallback scatter-transpose
    const int k  = t & 31;
    const int nb = t >> 5;
#pragma unroll
    for (int j = 0; j < 16; ++j)
      Bs[buf][(nb + j * 8) * LDB + k] = (_Float16)bRegF[j];
  };

  // ---- accumulators: 4 (M) x 2 (N) tiles of 16x16 -------------------------
  v8f acc[4][2] = {};

  auto compute = [&](int buf) {
    v16h a[4], b[2];
#pragma unroll
    for (int i = 0; i < 4; ++i) {
      // ISA 16-bit A layout: lane<16 holds K 0-7 & 16-23, lane>=16 K 8-15 & 24-31
      const _Float16* base = &As[buf][(wm * 64 + i * 16 + lmod) * LDA + lhalf * 8];
      v8h lo = *(const v8h*)base;
      v8h hi = *(const v8h*)(base + 16);
      a[i] = __builtin_shufflevector(lo, hi, 0,1,2,3,4,5,6,7,8,9,10,11,12,13,14,15);
    }
#pragma unroll
    for (int j = 0; j < 2; ++j) {
      // ISA 16-bit B layout: lane<16 holds K 0-7 & 16-23, lane>=16 K 8-15 & 24-31
      const _Float16* base = &Bs[buf][(wn * 32 + j * 16 + lmod) * LDB + lhalf * 8];
      v8h lo = *(const v8h*)base;
      v8h hi = *(const v8h*)(base + 16);
      b[j] = __builtin_shufflevector(lo, hi, 0,1,2,3,4,5,6,7,8,9,10,11,12,13,14,15);
    }
#pragma unroll
    for (int i = 0; i < 4; ++i)
#pragma unroll
      for (int j = 0; j < 2; ++j)
        acc[i][j] = __builtin_amdgcn_wmma_f32_16x16x32_f16(
            false, a[i], false, b[j], (short)0, acc[i][j], false, false);
  };

  // ---- prologue + single-barrier double-buffered K loop -------------------
  fetchA(0);
  if constexpr (WS) issueB(0, 0); else fetchB(0);
  storeA(0);
  if constexpr (!WS) storeB(0);

  int buf = 0;
  for (int kt = 0; kt < KTILES; ++kt) {
    if constexpr (WS)
      asm volatile("s_wait_asynccnt 0x0" ::: "memory");  // my DMAs into Bs[buf] done
    __syncthreads();   // publishes As[buf]/Bs[buf]; all waves done reading buf^1
    if (kt + 1 < KTILES) {
      fetchA(kt + 1);
      if constexpr (WS) issueB(kt + 1, buf ^ 1); else fetchB(kt + 1);
    }
    compute(buf);
    if (kt + 1 < KTILES) {
      storeA(buf ^ 1);
      if constexpr (!WS) storeB(buf ^ 1);
    }
    buf ^= 1;
  }

  // ---- epilogue: bias + SiLU, fp32 stores ---------------------------------
#pragma unroll
  for (int j = 0; j < 2; ++j) {
    const int n  = n0 + wn * 32 + j * 16 + lmod;   // each lane owns one column
    const float bv = bias[(size_t)n * NEXP + e];
#pragma unroll
    for (int i = 0; i < 4; ++i) {
      const size_t rowBase = tokBase + wm * 64 + i * 16 + 8 * lhalf;  // M = r + 8*(lane/16)
#pragma unroll
      for (int r = 0; r < 8; ++r) {
        float y = acc[i][j][r] + bv;
        float s = y / (1.0f + __expf(-y));         // silu(y) = y * sigmoid(y)
        out[(rowBase + r) * (size_t)D_OUT + n] = s;
      }
    }
  }
}

// ---------------------------------------------------------------------------
extern "C" void kernel_launch(void* const* d_in, const int* in_sizes, int n_in,
                              void* d_out, int out_size, void* d_ws, size_t ws_size,
                              hipStream_t stream) {
  const float* A    = (const float*)d_in[0];   // sorted_features fp32 [65536][1024]
  // d_in[1] = expert_ids_sorted (int32): equal static split per reference; unused.
  const float* W    = (const float*)d_in[2];   // routing_matrix fp32 [1024][2048][16]
  const float* bias = (const float*)d_in[3];   // routing_bias fp32 [2048][16]
  float*       out  = (float*)d_out;

  const size_t whBytes = (size_t)NEXP * D_IN * D_OUT * sizeof(_Float16);  // 64 MiB
  dim3 grid(D_OUT / BN, TOK_PER_E / BM, NEXP);   // (16, 32, 16)
  dim3 block(256);

  if (ws_size >= whBytes) {
    _Float16* Wh = (_Float16*)d_ws;
    dim3 pgrid(D_OUT / PN, D_IN / PK);           // (64, 16)
    moe_convert_w_kernel<<<pgrid, block, 0, stream>>>(W, Wh);
    moe_gemm_silu_kernel<true><<<grid, block, 0, stream>>>(A, nullptr, Wh, bias, out);
  } else {
    moe_gemm_silu_kernel<false><<<grid, block, 0, stream>>>(A, W, nullptr, bias, out);
  }
}
